// RNN_75118978007670
// MI455X (gfx1250) — matrix operand order repaired and checked by reading, not compiled, not run on hardware
//
#include <hip/hip_runtime.h>

#define T_STEPS 512
#define BATCH   128
#define DIN     256
#define DLAT    512

typedef __attribute__((ext_vector_type(16))) __bf16 v16bf;
typedef __attribute__((ext_vector_type(8)))  __bf16 v8bf;
typedef __attribute__((ext_vector_type(8)))  float  v8f;
typedef __attribute__((ext_vector_type(4)))  int    v4i;

union AFrag { v16bf v; v8bf h[2]; };

#define AS_GLOBAL __attribute__((address_space(1)))
#define AS_LDS    __attribute__((address_space(3)))

#if defined(__gfx1250__) && __has_builtin(__builtin_amdgcn_global_load_async_to_lds_b128)
#define XI_ASYNC 1
#else
#define XI_ASYNC 0
#endif

// Stage one contiguous 16x512 f32 tile (32 KB) into LDS: 8 x b128 per thread.
__device__ __forceinline__ void xi_stage(float* dst, const float* src, int tid) {
#pragma unroll
  for (int q = 0; q < 8; ++q) {
    const int e = (q * 256 + tid) * 4;  // float index, 16-byte granules
#if XI_ASYNC
    __builtin_amdgcn_global_load_async_to_lds_b128(
        (AS_GLOBAL v4i*)(src + e), (AS_LDS v4i*)(dst + e), 0, 0);
#else
    *(float4*)(dst + e) = *(const float4*)(src + e);
#endif
  }
}

__device__ __forceinline__ void xi_wait() {
#if XI_ASYNC
#if __has_builtin(__builtin_amdgcn_s_wait_asynccnt)
  __builtin_amdgcn_s_wait_asynccnt(0);
#else
  asm volatile("s_wait_asynccnt 0" ::: "memory");
#endif
#endif
}

// ---------------------------------------------------------------------------
// Kernel 1: weights -> bf16 (row-major == B-fragment feed order), fold biases.
// ---------------------------------------------------------------------------
__global__ __launch_bounds__(256) void rnn_prep(
    const float* __restrict__ Wi, const float* __restrict__ bi,
    const float* __restrict__ Wh, const float* __restrict__ bh,
    __bf16* __restrict__ Wib, __bf16* __restrict__ Whb,
    float* __restrict__ bias) {
  int i = blockIdx.x * 256 + threadIdx.x;
  if (i < DLAT * DLAT) Whb[i] = (__bf16)Wh[i];
  if (i < DLAT * DIN)  Wib[i] = (__bf16)Wi[i];
  if (i < DLAT)        bias[i] = bi[i] + bh[i];
}

// ---------------------------------------------------------------------------
// Kernel 2: Xi = X @ Wiᵀ + (bi+bh).  M-tile 16, N = 512 (8 waves x 4 tiles),
// K = 256.  Bulk-parallel (4096 blocks) so latency self-hides.
// ---------------------------------------------------------------------------
__global__ __launch_bounds__(256) void rnn_xproj(
    const float* __restrict__ X, const __bf16* __restrict__ Wib,
    const float* __restrict__ bias, float* __restrict__ Xi) {
  __shared__ __bf16 xs[16][DIN];
  const int tid = threadIdx.x;
  const int m0  = blockIdx.x * 16;

  for (int i = tid; i < 16 * DIN; i += 256)
    xs[i >> 8][i & (DIN - 1)] = (__bf16)X[(size_t)m0 * DIN + i];
  __syncthreads();

  const int lane = tid & 31, wave = tid >> 5;
  const int l16 = lane & 15, hi = lane >> 4;

  v8f acc[4];
#pragma unroll
  for (int i = 0; i < 4; ++i) {
    const int n0 = (wave * 4 + i) * 16;
    const float bv = bias[n0 + l16];
#pragma unroll
    for (int j = 0; j < 8; ++j) acc[i][j] = bv;
  }

  const __bf16* arow = &xs[l16][0];
#pragma unroll
  for (int kk = 0; kk < DIN / 32; ++kk) {
    const int k0 = kk * 32;
    AFrag a;
    a.h[0] = *(const v8bf*)(arow + k0 + hi * 16);
    a.h[1] = *(const v8bf*)(arow + k0 + 8 + hi * 16);
#pragma unroll
    for (int i = 0; i < 4; ++i) {
      const int n0 = (wave * 4 + i) * 16;
      v16bf b = *(const v16bf*)(Wib + (size_t)(n0 + l16) * DIN + k0 + hi * 16);
      acc[i] = __builtin_amdgcn_wmma_f32_16x16x32_bf16(
          false, a.v, false, b, (short)0, acc[i], false, false);
    }
  }

#pragma unroll
  for (int i = 0; i < 4; ++i) {
    const int n0 = (wave * 4 + i) * 16;
#pragma unroll
    for (int j = 0; j < 8; ++j)
      Xi[(size_t)(m0 + j + hi * 8) * DLAT + n0 + l16] = acc[i][j];
  }
}

// ---------------------------------------------------------------------------
// Kernel 3: batch-partitioned persistent scan.  Block g owns batch rows
// [16g,16g+16).  h double-buffered bf16 in LDS; Xi tiles streamed one step
// ahead by the async global->LDS unit; Wh B-fragments software-pipelined.
// One barrier per step (at step top).
// ---------------------------------------------------------------------------
__global__ __launch_bounds__(256) void rnn_scan(
    const float* __restrict__ Xi, const __bf16* __restrict__ Whb,
    const float* __restrict__ h0, float* __restrict__ H) {
  __shared__ __bf16 hbuf[2][16][DLAT];     // 32 KB
  __shared__ float  xibuf[2][16 * DLAT];   // 64 KB
  const int tid = threadIdx.x;
  const int r0  = blockIdx.x * 16;

  for (int i = tid; i < 16 * DLAT; i += 256)
    hbuf[0][i >> 9][i & (DLAT - 1)] = (__bf16)h0[(size_t)r0 * DLAT + i];

  xi_stage(&xibuf[0][0], Xi + (size_t)r0 * DLAT, tid);  // tile for t=0

  const int lane = tid & 31, wave = tid >> 5;
  const int l16 = lane & 15, hi = lane >> 4;
  const __bf16* wrow = Whb + (size_t)l16 * DLAT + hi * 16;  // per-lane B base

  for (int t = 0; t < T_STEPS; ++t) {
    const int cb = t & 1;
    xi_wait();          // this step's Xi tile landed in LDS (own lanes)
    __syncthreads();    // all lanes' async writes + prev step's hbuf writes

    // Prefetch next step's Xi tile while we compute.
    const int tn = (t + 1 < T_STEPS) ? t + 1 : t;
    xi_stage(&xibuf[cb ^ 1][0], Xi + ((size_t)tn * BATCH + r0) * DLAT, tid);

    // Seed accumulators from LDS (Xi already contains both biases).
    v8f acc[4];
#pragma unroll
    for (int i = 0; i < 4; ++i) {
      const int n0 = (wave * 4 + i) * 16;
#pragma unroll
      for (int j = 0; j < 8; ++j)
        acc[i][j] = xibuf[cb][(j + hi * 8) * DLAT + n0 + l16];
    }

    // Software-pipelined k-loop: fragments for kk+1 issue before WMMAs of kk.
    const __bf16* arow = &hbuf[cb][l16][0];
    AFrag a_cur, a_nxt;
    v16bf b_cur[4], b_nxt[4];
    a_cur.h[0] = *(const v8bf*)(arow + hi * 16);
    a_cur.h[1] = *(const v8bf*)(arow + 8 + hi * 16);
#pragma unroll
    for (int i = 0; i < 4; ++i)
      b_cur[i] = *(const v16bf*)(wrow + (size_t)((wave * 4 + i) * 16) * DLAT);

#pragma unroll
    for (int kk = 0; kk < DLAT / 32; ++kk) {
      if (kk < DLAT / 32 - 1) {
        const int k1 = (kk + 1) * 32;
        a_nxt.h[0] = *(const v8bf*)(arow + k1 + hi * 16);
        a_nxt.h[1] = *(const v8bf*)(arow + k1 + 8 + hi * 16);
#pragma unroll
        for (int i = 0; i < 4; ++i)
          b_nxt[i] = *(const v16bf*)(wrow + (size_t)((wave * 4 + i) * 16) * DLAT + k1);
      }
#pragma unroll
      for (int i = 0; i < 4; ++i)
        acc[i] = __builtin_amdgcn_wmma_f32_16x16x32_bf16(
            false, a_cur.v, false, b_cur[i], (short)0, acc[i], false, false);
      if (kk < DLAT / 32 - 1) {
        a_cur = a_nxt;
#pragma unroll
        for (int i = 0; i < 4; ++i) b_cur[i] = b_nxt[i];
      }
    }

    // h_t = tanh(.): fp32 to H, bf16 to next LDS buffer.
    const size_t obase = ((size_t)t * BATCH + r0) * DLAT;
#pragma unroll
    for (int i = 0; i < 4; ++i) {
      const int n0 = (wave * 4 + i) * 16;
#pragma unroll
      for (int j = 0; j < 8; ++j) {
        const int M = j + hi * 8;
        const float v = tanhf(acc[i][j]);
        H[obase + (size_t)M * DLAT + n0 + l16] = v;
        hbuf[cb ^ 1][M][n0 + l16] = (__bf16)v;
      }
    }
  }
}

// ---------------------------------------------------------------------------
extern "C" void kernel_launch(void* const* d_in, const int* in_sizes, int n_in,
                              void* d_out, int out_size, void* d_ws, size_t ws_size,
                              hipStream_t stream) {
  (void)in_sizes; (void)n_in; (void)out_size; (void)ws_size;
  const float* X  = (const float*)d_in[0];
  const float* h0 = (const float*)d_in[1];
  const float* Wi = (const float*)d_in[2];
  const float* bi = (const float*)d_in[3];
  const float* Wh = (const float*)d_in[4];
  const float* bh = (const float*)d_in[5];
  float* H = (float*)d_out;

  char* ws = (char*)d_ws;
  float* Xi = (float*)ws;
  size_t off = (size_t)T_STEPS * BATCH * DLAT * sizeof(float);   // 128 MB
  __bf16* Whb = (__bf16*)(ws + off); off += (size_t)DLAT * DLAT * 2;
  __bf16* Wib = (__bf16*)(ws + off); off += (size_t)DLAT * DIN * 2;
  float*  bias = (float*)(ws + off);

  rnn_prep<<<(DLAT * DLAT + 255) / 256, 256, 0, stream>>>(Wi, bi, Wh, bh, Wib, Whb, bias);
  rnn_xproj<<<(T_STEPS * BATCH) / 16, 256, 0, stream>>>(X, Wib, bias, Xi);
  rnn_scan<<<BATCH / 16, 256, 0, stream>>>(Xi, Whb, h0, H);
}
